// ContextualInvertedResidual_11751030522700
// MI455X (gfx1250) — matrix operand order repaired and checked by reading, not compiled, not run on hardware
//
#include <hip/hip_runtime.h>
#include <hip/hip_bf16.h>

// ---------------------------------------------------------------------------
// ContextualInvertedResidual on MI455X (gfx1250, wave32, WMMA bf16 16x16x32)
//
// N=32, C=64, E=384, H=W=56, HW=3136.
//  phase 1: h = relu6(bn1(Wexp @ x))   [GEMM 384x64 * 64xHW, bf16 WMMA]
//           ctx_sum[n][e] += sum_hw h  (LDS ds_add_f32 + one global atomic/ch)
//  phase 1.5: fb = (ctx_sum/HW @ Wctx^T)*s2 + t2   (tiny, pre-fused BN2 bias)
//  phase 2: t = relu6(dw3x3_scaled(h) + fb)        (bf16 tile in LDS)
//           out = bn3(Wproj @ t) + x               [GEMM 64x384 * 384xrow, WMMA]
// ---------------------------------------------------------------------------

typedef __attribute__((ext_vector_type(16))) __bf16         v16bf;
typedef __attribute__((ext_vector_type(16))) unsigned short ushort16;
typedef __attribute__((ext_vector_type(8)))  unsigned short ushort8;
typedef __attribute__((ext_vector_type(8)))  float          v8f;

#define NB   32
#define CIN  64
#define EC   384
#define HWD  3136
#define WID  56
#define TSTR 392   // padded t-tile row stride in shorts: 784B -> bank stride 4, 16B aligned

// bf16 helpers via bit ops (round-to-nearest-even)
__device__ __forceinline__ unsigned short f2bf(float f) {
  unsigned int u = __float_as_uint(f);
  u += 0x7FFFu + ((u >> 16) & 1u);
  return (unsigned short)(u >> 16);
}
__device__ __forceinline__ float bf2f(unsigned short s) {
  return __uint_as_float(((unsigned int)s) << 16);
}

union BFV { ushort16 us; v16bf bf; unsigned short s[16]; };
union F8V { v8f v; float f[8]; };

// ---------------------------------------------------------------------------
// prep: BN fold (scale = g/sqrt(v+eps), bias = b - m*scale) + zero ctx_sum
// ---------------------------------------------------------------------------
__global__ void prep_kernel(const float* g1, const float* b1, const float* m1, const float* v1,
                            const float* g2, const float* b2, const float* m2, const float* v2,
                            const float* g3, const float* b3, const float* m3, const float* v3,
                            float* s1, float* t1, float* s2, float* t2, float* s3, float* t3,
                            float* ctx_sum) {
  int i = blockIdx.x * blockDim.x + threadIdx.x;
  if (i < 384) {
    float sc = g1[i] / sqrtf(v1[i] + 1e-5f);
    s1[i] = sc; t1[i] = b1[i] - m1[i] * sc;
  } else if (i < 768) {
    int c = i - 384;
    float sc = g2[c] / sqrtf(v2[c] + 1e-5f);
    s2[c] = sc; t2[c] = b2[c] - m2[c] * sc;
  } else if (i < 832) {
    int c = i - 768;
    float sc = g3[c] / sqrtf(v3[c] + 1e-5f);
    s3[c] = sc; t3[c] = b3[c] - m3[c] * sc;
  }
  if (i < NB * EC) ctx_sum[i] = 0.0f;
}

// ---------------------------------------------------------------------------
// swizzleA: f32 weight [M,K] row-major -> bf16 WMMA A fragments
// fragment (mt,kt): 32 lanes x 16 bf16, contiguous-K A 16x32 layout:
//   lane 0-15  row mt*16+lane : K = kt*32 + {0..15}
//   lane 16-31 row mt*16+l-16 : K = kt*32 + {16..31}
// ---------------------------------------------------------------------------
__global__ void swizzleA_kernel(const float* __restrict__ w, unsigned short* __restrict__ dst,
                                int Kt, int K, int total) {
  int i = blockIdx.x * blockDim.x + threadIdx.x;
  if (i >= total) return;
  int j    = i & 15;
  int lane = (i >> 4) & 31;
  int rest = i >> 9;
  int kt   = rest % Kt;
  int mt   = rest / Kt;
  int hf   = lane >> 4;
  int k    = kt * 32 + hf * 16 + j;
  int row  = mt * 16 + (lane & 15);
  dst[i] = f2bf(w[row * K + k]);
}

// ---------------------------------------------------------------------------
// expand: per block = (pixel-tile of 16, sample n). 8 waves, 3 M-tiles each.
// Pooled context accumulated per block in LDS with ds_add_f32 (no-return,
// fire-and-forget), flushed once per channel with global_atomic_add_f32.
// ---------------------------------------------------------------------------
__global__ __launch_bounds__(256) void expand_kernel(
    const float* __restrict__ x, const v16bf* __restrict__ Aexp,
    const float* __restrict__ s1, const float* __restrict__ t1,
    unsigned short* __restrict__ h, float* __restrict__ ctx_sum) {
  __shared__ float xs[CIN * 16];
  __shared__ float s1s[EC], t1s[EC], csum[EC];
  const int n = blockIdx.y, pbase = blockIdx.x * 16, tid = threadIdx.x;

  for (int i = tid; i < CIN * 16; i += 256) {
    int c = i >> 4, p = i & 15;
    xs[i] = x[(n * CIN + c) * HWD + pbase + p];
  }
  for (int i = tid; i < EC; i += 256) {
    s1s[i] = s1[i]; t1s[i] = t1[i]; csum[i] = 0.0f;
  }
  __syncthreads();

  const int lane = tid & 31, wave = tid >> 5;
  const int hf = lane >> 4, pcol = lane & 15;

  // B fragments (x tile), contiguous-K B 32x16 layout (same k order as A): lane half <-> K 0-15/16-31
  BFV b0u, b1u;
  for (int j = 0; j < 16; ++j) {
    b0u.s[j] = f2bf(xs[(hf * 16 + j) * 16 + pcol]);
    b1u.s[j] = f2bf(xs[(32 + hf * 16 + j) * 16 + pcol]);
  }
  const v16bf b0 = b0u.bf, b1 = b1u.bf;

  for (int mt = wave; mt < 24; mt += 8) {
    v16bf a0 = Aexp[(mt * 2 + 0) * 32 + lane];
    v16bf a1 = Aexp[(mt * 2 + 1) * 32 + lane];
    v8f acc = {};
    acc = __builtin_amdgcn_wmma_f32_16x16x32_bf16(false, a0, false, b0, (short)0, acc, false, false);
    acc = __builtin_amdgcn_wmma_f32_16x16x32_bf16(false, a1, false, b1, (short)0, acc, false, false);
    F8V r8; r8.v = acc;
    #pragma unroll
    for (int r = 0; r < 8; ++r) {
      int ch = mt * 16 + hf * 8 + r;                 // C/D layout: VGPR r, halves M=r / M=r+8
      float v = r8.f[r] * s1s[ch] + t1s[ch];
      v = fminf(fmaxf(v, 0.0f), 6.0f);               // ReLU6
      h[(n * EC + ch) * HWD + pbase + pcol] = f2bf(v);
      atomicAdd(&csum[ch], v);                       // ds_add_f32, banks spread over ch%64
    }
  }
  __syncthreads();
  for (int i = tid; i < EC; i += 256)
    atomicAdd(&ctx_sum[n * EC + i], csum[i]);        // global_atomic_add_f32
}

// ---------------------------------------------------------------------------
// ctx/fb: fb[n][o] = ((1/HW) * sum_c ctx_sum[n][c] * w_ctx[o][c]) * s2[o] + t2[o]
// (folds the BN2 bias path so phase 2 does a single fused add)
// ---------------------------------------------------------------------------
__global__ void ctxbias_kernel(const float* __restrict__ ctx_sum,
                               const float* __restrict__ w_ctx,
                               const float* __restrict__ s2,
                               const float* __restrict__ t2,
                               float* __restrict__ fb) {
  int i = blockIdx.x * blockDim.x + threadIdx.x;
  if (i >= NB * EC) return;
  int n = i / EC, o = i - n * EC;
  const float* cs = ctx_sum + n * EC;
  const float* wr = w_ctx + o * EC;
  float s = 0.0f;
  for (int c = 0; c < EC; ++c) s += cs[c] * wr[c];
  fb[i] = s * (1.0f / (float)HWD) * s2[o] + t2[o];
}

// ---------------------------------------------------------------------------
// dwproj: per block = (row y, sample n). dw3x3 with BN2-scale folded into the
// LDS-staged taps, + fused bias fb, relu6 -> bf16 LDS tile t[p][e] (row stride
// padded to 392 shorts: 16B-aligned fragments, near-conflict-free stores),
// then project GEMM via WMMA, bn3 + residual, f32 store.
// ---------------------------------------------------------------------------
__global__ __launch_bounds__(256) void dwproj_kernel(
    const float* __restrict__ x, const unsigned short* __restrict__ h,
    const float* __restrict__ w_dw, const float* __restrict__ s2,
    const float* __restrict__ fb,
    const v16bf* __restrict__ Aproj,
    const float* __restrict__ s3, const float* __restrict__ t3,
    float* __restrict__ out) {
  __shared__ float          wdw_s[EC * 9];       // 13.5 KB (pre-scaled by s2)
  __shared__ unsigned short t_s[64 * TSTR];      // 50176 B, total 64000 B LDS
  const int y = blockIdx.x, n = blockIdx.y, tid = threadIdx.x;

  // Stage depthwise taps with BN2 scale folded in.
  for (int i = tid; i < EC * 9; i += 256) wdw_s[i] = w_dw[i] * s2[i / 9];

  // Warm L1/L2 for the three bf16 h-rows this block gathers from (~129 KB).
  {
    const unsigned short* hrow = h + (n * EC) * HWD;
    int y0 = (y > 0) ? (y - 1) : 0;
    int y1 = (y < WID - 1) ? (y + 1) : (WID - 1);
    int nrows = y1 - y0 + 1;
    // one prefetch per 32 channels-elements (64B) per row segment
    for (int i = tid; i < EC * nrows * 2; i += 256) {
      int rr = i % (nrows * 2);
      int e  = i / (nrows * 2);
      int yy = y0 + (rr >> 1);
      int seg = rr & 1;
      __builtin_prefetch(hrow + e * HWD + yy * WID + seg * 32, 0, 1);
    }
  }
  __syncthreads();

  for (int i = tid; i < EC * 64; i += 256) {
    int e = i >> 6, xx = i & 63;
    float v = 0.0f;
    if (xx < WID) {
      float acc = 0.0f;
      const unsigned short* hb = h + (n * EC + e) * HWD;
      const float* wk = &wdw_s[e * 9];
      #pragma unroll
      for (int dy = -1; dy <= 1; ++dy) {
        int yy = y + dy;
        if (yy < 0 || yy >= WID) continue;
        #pragma unroll
        for (int dx = -1; dx <= 1; ++dx) {
          int xc = xx + dx;
          if (xc < 0 || xc >= WID) continue;
          acc += bf2f(hb[yy * WID + xc]) * wk[(dy + 1) * 3 + (dx + 1)];
        }
      }
      v = acc + fb[n * EC + e];                  // dw*s2 + (cb*s2 + t2)
      v = fminf(fmaxf(v, 0.0f), 6.0f);
    }
    t_s[xx * TSTR + e] = f2bf(v);                // zero-padded for p >= 56
  }
  __syncthreads();

  const int lane = tid & 31, wave = tid >> 5;
  const int hf = lane >> 4, pcol = lane & 15;

  for (int wt = wave; wt < 16; wt += 8) {        // 4 M-tiles x 4 N-tiles
    int mt = wt & 3, nt = wt >> 2;
    const unsigned short* brow = &t_s[(nt * 16 + pcol) * TSTR + hf * 16];
    v8f acc = {};
    #pragma unroll
    for (int ks = 0; ks < 12; ++ks) {            // K = 384 = 12 * 32
      v16bf a = Aproj[(mt * 12 + ks) * 32 + lane];
      BFV b;                                     // two 16B-aligned ds_load_b128
      *(ushort8*)&b.s[0] = *(const ushort8*)&brow[ks * 32];
      *(ushort8*)&b.s[8] = *(const ushort8*)&brow[ks * 32 + 8];
      acc = __builtin_amdgcn_wmma_f32_16x16x32_bf16(false, a, false, b.bf, (short)0, acc, false, false);
    }
    int p = nt * 16 + pcol;
    if (p < WID) {
      F8V r8; r8.v = acc;
      #pragma unroll
      for (int r = 0; r < 8; ++r) {
        int o = mt * 16 + hf * 8 + r;
        int idx = (n * CIN + o) * HWD + y * WID + p;
        out[idx] = r8.f[r] * s3[o] + t3[o] + x[idx];
      }
    }
  }
}

// ---------------------------------------------------------------------------
// workspace layout (bytes, 256B aligned)
// ---------------------------------------------------------------------------
#define OFF_AEXP    0u            // 24*2*512 bf16  = 49152
#define OFF_APROJ   49152u        // 4*12*512 bf16  = 49152
#define OFF_S1      98304u
#define OFF_T1      99840u
#define OFF_S2      101376u
#define OFF_T2      102912u
#define OFF_S3      104448u
#define OFF_T3      104704u
#define OFF_CTXSUM  104960u       // 32*384 f32 = 49152
#define OFF_CTXBIAS 154112u       // 32*384 f32 = 49152 (fused fb)
#define OFF_H       203264u       // 32*384*3136 bf16 = 77070336  -> total ~77.3 MB

extern "C" void kernel_launch(void* const* d_in, const int* in_sizes, int n_in,
                              void* d_out, int out_size, void* d_ws, size_t ws_size,
                              hipStream_t stream) {
  const float* x        = (const float*)d_in[0];
  const float* w_expand = (const float*)d_in[1];
  const float* g1 = (const float*)d_in[2];
  const float* b1 = (const float*)d_in[3];
  const float* m1 = (const float*)d_in[4];
  const float* v1 = (const float*)d_in[5];
  const float* w_dw  = (const float*)d_in[6];
  const float* w_ctx = (const float*)d_in[7];
  const float* g2 = (const float*)d_in[8];
  const float* b2 = (const float*)d_in[9];
  const float* m2 = (const float*)d_in[10];
  const float* v2 = (const float*)d_in[11];
  const float* w_proj = (const float*)d_in[12];
  const float* g3 = (const float*)d_in[13];
  const float* b3 = (const float*)d_in[14];
  const float* m3 = (const float*)d_in[15];
  const float* v3 = (const float*)d_in[16];

  char* ws = (char*)d_ws;
  unsigned short* aexp  = (unsigned short*)(ws + OFF_AEXP);
  unsigned short* aproj = (unsigned short*)(ws + OFF_APROJ);
  float* s1 = (float*)(ws + OFF_S1); float* t1 = (float*)(ws + OFF_T1);
  float* s2 = (float*)(ws + OFF_S2); float* t2 = (float*)(ws + OFF_T2);
  float* s3 = (float*)(ws + OFF_S3); float* t3 = (float*)(ws + OFF_T3);
  float* ctx_sum = (float*)(ws + OFF_CTXSUM);
  float* fb      = (float*)(ws + OFF_CTXBIAS);
  unsigned short* hbuf = (unsigned short*)(ws + OFF_H);

  prep_kernel<<<48, 256, 0, stream>>>(g1, b1, m1, v1, g2, b2, m2, v2, g3, b3, m3, v3,
                                      s1, t1, s2, t2, s3, t3, ctx_sum);
  swizzleA_kernel<<<48, 256, 0, stream>>>(w_expand, aexp, 2, 64, 24576);
  swizzleA_kernel<<<48, 256, 0, stream>>>(w_proj, aproj, 12, 384, 24576);

  expand_kernel<<<dim3(HWD / 16, NB), 256, 0, stream>>>(
      x, (const v16bf*)aexp, s1, t1, hbuf, ctx_sum);

  ctxbias_kernel<<<48, 256, 0, stream>>>(ctx_sum, w_ctx, s2, t2, fb);

  dwproj_kernel<<<dim3(WID, NB), 256, 0, stream>>>(
      x, hbuf, w_dw, s2, fb, (const v16bf*)aproj, s3, t3, (float*)d_out);
}
